// _TopK_23227183137544
// MI455X (gfx1250) — matrix operand order repaired and checked by reading, not compiled, not run on hardware
//
#include <hip/hip_runtime.h>

#define DIMF 128
#define BN_EPS 1e-5f

typedef __attribute__((ext_vector_type(16))) __bf16 v16bf;
typedef __attribute__((ext_vector_type(8)))  __bf16 v8bf;
typedef __attribute__((ext_vector_type(8)))  float  v8f;

// ---------------- small utility kernels ----------------

__global__ void k_zero(float* __restrict__ p, int n) {
    int i = blockIdx.x * blockDim.x + threadIdx.x;
    if (i < n) p[i] = 0.0f;
}

__global__ void k_degree(const int* __restrict__ src, const int* __restrict__ dst,
                         float* __restrict__ deg_out, float* __restrict__ deg_in,
                         int n_edges) {
    int i = blockIdx.x * blockDim.x + threadIdx.x;
    if (i < n_edges) {
        atomicAdd(&deg_out[src[i]], 1.0f);
        atomicAdd(&deg_in[dst[i]], 1.0f);
    }
}

// in-place: deg -> rsqrt(deg + 1)   (+1 = self loop; always >= 1 so max() is moot)
__global__ void k_inv_sqrt(float* __restrict__ dout, float* __restrict__ din, int n) {
    int i = blockIdx.x * blockDim.x + threadIdx.x;
    if (i < n) {
        dout[i] = rsqrtf(dout[i] + 1.0f);
        din[i]  = rsqrtf(din[i]  + 1.0f);
    }
}

// Wt[l][n][k] = W[l][k][n], converted to bf16 (B operand wants K-contiguous per column)
__global__ void k_convert_wt(const float* __restrict__ W, __bf16* __restrict__ Wt, int total) {
    int t = blockIdx.x * blockDim.x + threadIdx.x;
    if (t < total) {
        int l = t >> 14, rem = t & 16383, nn = rem >> 7, kk = rem & 127;
        Wt[t] = (__bf16)W[(l << 14) + (kk << 7) + nn];
    }
}

// layer-0 prep: copy feat into output slot 0 and init agg with the self-loop term
__global__ void k_init0(const float* __restrict__ feat, const float* __restrict__ inv_out,
                        float* __restrict__ out0, float* __restrict__ agg, int nd) {
    int i = blockIdx.x * blockDim.x + threadIdx.x;
    if (i < nd) {
        float f = feat[i];
        out0[i] = f;
        agg[i]  = f * inv_out[i >> 7];
    }
}

// ---------------- SpMM: agg[dst] += h[src] * inv_out[src] ----------------
// one wave per edge; src/dst are wave-uniform -> force them into SGPRs so the
// row gather and the 4 f32 atomics use saddr-form addressing.

__global__ void k_spmm(const float* __restrict__ h, const int* __restrict__ src,
                       const int* __restrict__ dst, const float* __restrict__ inv_out,
                       float* __restrict__ agg, int n_edges) {
    int g = blockIdx.x * blockDim.x + threadIdx.x;
    int e = g >> 5;
    if (e >= n_edges) return;
    int j = g & 31;
    int s = __builtin_amdgcn_readfirstlane(src[e]);
    int d = __builtin_amdgcn_readfirstlane(dst[e]);
    float sc = inv_out[s];
    float4 hv = reinterpret_cast<const float4*>(h + (size_t)s * DIMF)[j];
    float* o = agg + (size_t)d * DIMF + (j << 2);
    atomicAdd(o + 0, hv.x * sc);
    atomicAdd(o + 1, hv.y * sc);
    atomicAdd(o + 2, hv.z * sc);
    atomicAdd(o + 3, hv.w * sc);
}

// ---------------- GEMM: z = (agg * inv_in) @ W + b, in-place over agg ----------------
// block = 256 threads = 8 waves; block owns 16 rows, wave w owns column tile [16w,16w+16)
// A tile staged (scaled + bf16) in LDS; 4x v_wmma_f32_16x16x32_bf16 per wave.

__global__ void __launch_bounds__(256)
k_gemm(float* __restrict__ zio,            // in: agg rows, out: z rows (in-place safe)
       const __bf16* __restrict__ Wt,      // [128 cols][128 k]
       const float* __restrict__ bias,     // [128]
       const float* __restrict__ inv_in,   // [n]
       int n_rows) {
    __shared__ __bf16 As[16][136];         // pad 128->136 to dodge bank conflicts
    const int r0 = blockIdx.x << 4;
    const int t  = threadIdx.x;

    // cooperative stage: 16x128 f32 -> row-scaled bf16 in LDS
#pragma unroll
    for (int i = 0; i < 2; ++i) {
        int v    = t + (i << 8);           // 0..511 float4 slots
        int row  = v >> 5;
        int k4   = v & 31;
        int grow = r0 + row;
        float4 val = {0.f, 0.f, 0.f, 0.f};
        if (grow < n_rows) {
            val = reinterpret_cast<const float4*>(zio)[grow * 32 + k4];
            float s = inv_in[grow];
            val.x *= s; val.y *= s; val.z *= s; val.w *= s;
        }
        __bf16* dp = &As[row][k4 << 2];
        dp[0] = (__bf16)val.x; dp[1] = (__bf16)val.y;
        dp[2] = (__bf16)val.z; dp[3] = (__bf16)val.w;
    }
    __syncthreads();                       // after this, all global reads of our rows are done

    const int wave = t >> 5;
    const int lane = t & 31;
    const int nloc = lane & 15;
    const int hi   = lane >> 4;            // half-wave selector
    const int col  = (wave << 4) + nloc;

    // B fragments: lane holds column `col`; K = kb*32 + hi*8 + [0..7] and +16 (ISA B table)
    v16bf bfrag[4];
    const __bf16* wcol = Wt + (col << 7) + (hi << 3);
#pragma unroll
    for (int kb = 0; kb < 4; ++kb) {
        const v8bf* p = reinterpret_cast<const v8bf*>(wcol + (kb << 5));
        v8bf b0 = p[0], b1 = p[2];
#pragma unroll
        for (int q = 0; q < 8; ++q) { bfrag[kb][q] = b0[q]; bfrag[kb][q + 8] = b1[q]; }
    }

    // A fragments: lane row = nloc; K = kb*32 + hi*8 + [0..7] and +16 (ISA A table)
    v8f acc = {};
#pragma unroll
    for (int kb = 0; kb < 4; ++kb) {
        const __bf16* ab = &As[nloc][(kb << 5) + (hi << 3)];
        v8bf a0 = *reinterpret_cast<const v8bf*>(ab);
        v8bf a1 = *reinterpret_cast<const v8bf*>(ab + 16);
        v16bf afrag;
#pragma unroll
        for (int q = 0; q < 8; ++q) { afrag[q] = a0[q]; afrag[q + 8] = a1[q]; }
        acc = __builtin_amdgcn_wmma_f32_16x16x32_bf16(false, afrag, false, bfrag[kb],
                                                      (short)0, acc, false, false);
    }

    // C/D layout: lanes 0-15 -> M = vgpr 0..7, lanes 16-31 -> M = 8..15, N = lane%16
    const float bv  = bias[col];
    const int rbase = r0 + (hi << 3);
    if (r0 + 16 <= n_rows) {
        // block-uniform fast path (always taken when n_rows % 16 == 0): no EXEC churn
#pragma unroll
        for (int q = 0; q < 8; ++q)
            zio[(rbase + q) * DIMF + col] = acc[q] + bv;
    } else {
#pragma unroll
        for (int q = 0; q < 8; ++q) {
            int row = rbase + q;
            if (row < n_rows) zio[row * DIMF + col] = acc[q] + bv;
        }
    }
}

// ---------------- BatchNorm ----------------

__global__ void k_bn_reduce(const float* __restrict__ z, float* __restrict__ sums,
                            float* __restrict__ sqs, int n_rows) {
    int g = blockIdx.x * blockDim.x + threadIdx.x;
    int c = g & 127;
    int r = g >> 7;
    int stride = (gridDim.x * blockDim.x) >> 7;
    float s = 0.f, q = 0.f;
    for (; r < n_rows; r += stride) {
        float v = z[r * DIMF + c];
        s += v; q += v * v;
    }
    atomicAdd(&sums[c], s);
    atomicAdd(&sqs[c], q);
}

__global__ void k_bn_finalize(const float* __restrict__ sums, const float* __restrict__ sqs,
                              const float* __restrict__ gamma, const float* __restrict__ beta,
                              float* __restrict__ scale, float* __restrict__ shift, int n_rows) {
    int c = threadIdx.x;
    if (c < DIMF) {
        float invn = 1.0f / (float)n_rows;
        float mean = sums[c] * invn;
        float var  = sqs[c] * invn - mean * mean;   // biased variance (matches jnp.var)
        float sc   = gamma[c] * rsqrtf(var + BN_EPS);
        scale[c] = sc;
        shift[c] = beta[c] - mean * sc;
    }
}

// h = relu(z*scale + shift) -> output slot; agg = h * inv_out (next layer's self-loop init)
__global__ void k_bn_apply(const float* __restrict__ z, const float* __restrict__ scale,
                           const float* __restrict__ shift, const float* __restrict__ inv_out,
                           float* __restrict__ hout, float* __restrict__ aggout, int nd) {
    int i = blockIdx.x * blockDim.x + threadIdx.x;
    if (i < nd) {
        int c = i & 127;
        float v = fmaxf(z[i] * scale[c] + shift[c], 0.0f);
        hout[i]   = v;
        aggout[i] = v * inv_out[i >> 7];
    }
}

// ---------------- launch ----------------

extern "C" void kernel_launch(void* const* d_in, const int* in_sizes, int n_in,
                              void* d_out, int out_size, void* d_ws, size_t ws_size,
                              hipStream_t stream) {
    const float* feat   = (const float*)d_in[0];
    const float* Ws     = (const float*)d_in[1];
    const float* bs     = (const float*)d_in[2];
    const float* gammas = (const float*)d_in[3];
    const float* betas  = (const float*)d_in[4];
    const int*   esrc   = (const int*)d_in[5];
    const int*   edst   = (const int*)d_in[6];

    const int n        = in_sizes[0] / DIMF;
    const int n_edges  = in_sizes[5];
    const int n_layers = in_sizes[1] / (DIMF * DIMF);
    const int nd       = n * DIMF;

    float* out = (float*)d_out;

    // workspace carve-up (~52 MB): inv_out | inv_in | agg(=z in-place) | Wt(bf16) | BN stats
    float*  inv_out = (float*)d_ws;
    float*  inv_in  = inv_out + n;
    float*  agg     = inv_in + n;
    __bf16* Wt      = (__bf16*)(agg + (size_t)nd);
    float*  bstat   = (float*)(Wt + (size_t)n_layers * DIMF * DIMF);
    float*  bsum = bstat, *bsq = bstat + 128, *bscale = bstat + 256, *bshift = bstat + 384;

    const int T = 256;

    // degrees -> inv sqrt (inv_out/inv_in are contiguous: one zero pass)
    k_zero<<<(2 * n + T - 1) / T, T, 0, stream>>>(inv_out, 2 * n);
    k_degree<<<(n_edges + T - 1) / T, T, 0, stream>>>(esrc, edst, inv_out, inv_in, n_edges);
    k_inv_sqrt<<<(n + T - 1) / T, T, 0, stream>>>(inv_out, inv_in, n);

    const int wtot = n_layers * DIMF * DIMF;
    k_convert_wt<<<(wtot + T - 1) / T, T, 0, stream>>>(Ws, Wt, wtot);

    k_init0<<<(nd + T - 1) / T, T, 0, stream>>>(feat, inv_out, out, agg, nd);

    for (int l = 0; l < n_layers; ++l) {
        const float* h     = out + (size_t)l * nd;
        float*       hnext = out + (size_t)(l + 1) * nd;

        long tot = (long)n_edges * 32;
        k_spmm<<<(int)((tot + T - 1) / T), T, 0, stream>>>(h, esrc, edst, inv_out, agg, n_edges);

        k_gemm<<<(n + 15) / 16, T, 0, stream>>>(agg, Wt + (size_t)l * DIMF * DIMF,
                                                bs + l * DIMF, inv_in, n);

        k_zero<<<1, 256, 0, stream>>>(bsum, 256);   // sums + sqs
        k_bn_reduce<<<512, T, 0, stream>>>(agg, bsum, bsq, n);
        k_bn_finalize<<<1, 128, 0, stream>>>(bsum, bsq, gammas + l * DIMF, betas + l * DIMF,
                                             bscale, bshift, n);
        k_bn_apply<<<(nd + T - 1) / T, T, 0, stream>>>(agg, bscale, bshift, inv_out,
                                                       hnext, agg, nd);
    }
}
